// AttentionBlock_26740466385092
// MI455X (gfx1250) — matrix operand order repaired and checked by reading, not compiled, not run on hardware
//
#include <hip/hip_runtime.h>
#include <math.h>

typedef __bf16 bf16;
typedef __attribute__((ext_vector_type(16))) __bf16 v16bf;
typedef __attribute__((ext_vector_type(8)))  __bf16 v8bf;
typedef __attribute__((ext_vector_type(8)))  float  v8f;

#define CCH   128          // channels
#define NPIX  4096         // 64*64 spatial
#define NB    4            // batch
#define KB    64           // key block for flash attention
#define QSCALE 0.08838834764831845f   // 128^-0.5

// ---------- bf16 conversion: native cast (RNE), lowers to HW cvt ----------
static __device__ __forceinline__ bf16 f2bf(float f) {
    return (bf16)f;
}

static __device__ __forceinline__ v16bf cat8(v8bf lo, v8bf hi) {
    return __builtin_shufflevector(lo, hi, 0,1,2,3,4,5,6,7,8,9,10,11,12,13,14,15);
}

// A-fragment: 16x32 (MxK) bf16, memory row-major [M][ld].
// lane<16 : row=lane,    K = k0 + {0..7, 16..23}
// lane>=16: row=lane-16, K = k0 + {8..15, 24..31}
static __device__ __forceinline__ v16bf loadA(const bf16* base, int ld,
                                              int row0, int k0, int lane) {
    int l16 = lane & 15, half = lane >> 4;
    const bf16* p = base + (size_t)(row0 + l16) * ld + k0 + half * 8;
    v8bf lo = *reinterpret_cast<const v8bf*>(p);
    v8bf hi = *reinterpret_cast<const v8bf*>(p + 16);
    return cat8(lo, hi);
}

// B-fragment: 32x16 (KxN) bf16, memory is B^T row-major: [N][ld] (row = col).
// lane<16 : N=lane,    K = k0 + {0..7, 16..23}
// lane>=16: N=lane-16, K = k0 + {8..15, 24..31}
static __device__ __forceinline__ v16bf loadB(const bf16* base, int ld,
                                              int col0, int k0, int lane) {
    int l16 = lane & 15, half = lane >> 4;
    const bf16* p = base + (size_t)(col0 + l16) * ld + k0 + half * 8;
    v8bf lo = *reinterpret_cast<const v8bf*>(p);
    v8bf hi = *reinterpret_cast<const v8bf*>(p + 16);
    return cat8(lo, hi);
}

static __device__ __forceinline__ v8f wmma_bf16(v16bf a, v16bf b, v8f c) {
    return __builtin_amdgcn_wmma_f32_16x16x32_bf16(false, a, false, b,
                                                   (short)0, c, false, false);
}

// Async DMA 64B (4x16B) from global to LDS; tracked by ASYNCcnt.
// IOFFSET is added to BOTH the global and LDS addresses (ISA 15.18.3),
// so one base pair + offset immediates covers the whole chunk.
static __device__ __forceinline__ void async_copy64(const bf16* gsrc, bf16* ldst) {
    unsigned lds = (unsigned)(size_t)ldst;     // generic LDS ptr: low 32b = LDS addr
    unsigned long long ga = (unsigned long long)(size_t)gsrc;
    asm volatile(
        "global_load_async_to_lds_b128 %0, %1, off\n\t"
        "global_load_async_to_lds_b128 %0, %1, off offset:16\n\t"
        "global_load_async_to_lds_b128 %0, %1, off offset:32\n\t"
        "global_load_async_to_lds_b128 %0, %1, off offset:48"
        :: "v"(lds), "v"(ga) : "memory");
}

static __device__ __forceinline__ void wait_async0() {
    asm volatile("s_wait_asynccnt 0x0" ::: "memory");
}

// ============ kernel 0: weights f32 -> bf16 (q,k,v,p packed) ============
__global__ __launch_bounds__(256) void wcvt_kernel(const float* __restrict__ wq,
                                                   const float* __restrict__ wk,
                                                   const float* __restrict__ wv,
                                                   const float* __restrict__ wp,
                                                   bf16* __restrict__ wbf) {
    int i = blockIdx.x * 256 + threadIdx.x;          // 0 .. 65535
    const float* src = (i < 16384) ? wq : (i < 32768) ? wk : (i < 49152) ? wv : wp;
    wbf[i] = f2bf(src[i & 16383]);
}

// ============ kernel 1: group-norm statistics ============
__global__ __launch_bounds__(256) void gn_stats_kernel(const float* __restrict__ x,
                                                       float* __restrict__ stats) {
    int g = blockIdx.x, b = blockIdx.y;
    const float* xb = x + ((size_t)b * CCH + g * 16) * NPIX;
    float s = 0.f, s2 = 0.f;
    for (int i = threadIdx.x; i < 16 * NPIX; i += 256) {
        float v = xb[i];
        s += v; s2 += v * v;
    }
    for (int d = 16; d >= 1; d >>= 1) {
        s  += __shfl_xor(s,  d);
        s2 += __shfl_xor(s2, d);
    }
    __shared__ float rs[8], rs2[8];
    int lane = threadIdx.x & 31, wave = threadIdx.x >> 5;
    if (lane == 0) { rs[wave] = s; rs2[wave] = s2; }
    __syncthreads();
    if (threadIdx.x == 0) {
        float S = 0.f, S2 = 0.f;
        for (int i = 0; i < 8; i++) { S += rs[i]; S2 += rs2[i]; }
        float mean = S * (1.f / 65536.f);
        float var  = S2 * (1.f / 65536.f) - mean * mean;
        stats[(b * 8 + g) * 2 + 0] = mean;
        stats[(b * 8 + g) * 2 + 1] = rsqrtf(var + 1e-5f);
    }
}

// ============ kernel 2: fused groupnorm + QKV projection (WMMA) ============
// grid (32 pixel-tiles, 4 batch), 256 threads (8 waves); wave w owns pixel
// sub-tile w (16 px): B-fragments loaded once, reused for all 24 output tiles.
__global__ __launch_bounds__(256) void qkv_kernel(const float* __restrict__ x,
                                                  const float* __restrict__ gamma,
                                                  const float* __restrict__ beta,
                                                  const bf16*  __restrict__ wbf,
                                                  const float* __restrict__ bq,
                                                  const float* __restrict__ bk,
                                                  const float* __restrict__ bv,
                                                  const float* __restrict__ stats,
                                                  bf16* __restrict__ qws,
                                                  bf16* __restrict__ kws,
                                                  bf16* __restrict__ vws) {
    __shared__ bf16 xs[128][CCH + 8];      // normalized tile, [pixel][channel]
    __shared__ float sm[8], sr[8];
    int b = blockIdx.y, pblk = blockIdx.x, tid = threadIdx.x;
    if (tid < 8) {
        sm[tid] = stats[(b * 8 + tid) * 2 + 0];
        sr[tid] = stats[(b * 8 + tid) * 2 + 1];
    }
    __syncthreads();
    int pbase = pblk * 128;
    const float* xb = x + (size_t)b * CCH * NPIX;
    for (int i = tid; i < 128 * 128; i += 256) {
        int c = i >> 7, p = i & 127;
        float v  = xb[(size_t)c * NPIX + pbase + p];
        int g    = c >> 4;
        float hn = (v - sm[g]) * sr[g] * gamma[c] + beta[c];
        xs[p][c] = f2bf(hn);
    }
    __syncthreads();

    int lane = tid & 31, wave = tid >> 5;
    int half = lane >> 4, l16 = lane & 15;

    v16bf bb[4];                                   // invariant B fragments
    #pragma unroll
    for (int i = 0; i < 4; i++) bb[i] = loadB(&xs[0][0], CCH + 8, wave * 16, i * 32, lane);

    int p = pbase + wave * 16 + l16;
    #pragma unroll
    for (int mat = 0; mat < 3; mat++) {
        const bf16*  W    = wbf + (size_t)mat * 16384;
        const float* bias = (mat == 0) ? bq : (mat == 1) ? bk : bv;
        bf16*        dst  = ((mat == 0) ? qws : (mat == 1) ? kws : vws)
                            + (size_t)b * NPIX * CCH;
        float sc = (mat == 0) ? QSCALE : 1.0f;
        for (int ot = 0; ot < 8; ot++) {
            v8f acc; for (int r = 0; r < 8; r++) acc[r] = 0.f;
            #pragma unroll
            for (int i = 0; i < 4; i++)
                acc = wmma_bf16(loadA(W, CCH, ot * 16, i * 32, lane), bb[i], acc);
            int o0 = ot * 16 + half * 8;
            v8f bvv = *reinterpret_cast<const v8f*>(&bias[o0]);   // 32B load
            v8bf pk;
            #pragma unroll
            for (int r = 0; r < 8; r++) pk[r] = f2bf((acc[r] + bvv[r]) * sc);
            *reinterpret_cast<v8bf*>(&dst[(size_t)p * CCH + o0]) = pk;   // 16B store
        }
    }
}

// ============ kernel 3: flash attention (WMMA + online softmax) ============
// grid (32 query-tiles of 128, 4 batch); wave w owns 16 query rows; KB=64 keys
__global__ __launch_bounds__(256) void attn_kernel(const bf16* __restrict__ qws,
                                                   const bf16* __restrict__ kws,
                                                   const bf16* __restrict__ vws,
                                                   bf16* __restrict__ aws) {
    __shared__ bf16 Kblk[KB][CCH + 8];     // [key][channel]
    __shared__ bf16 Vt[CCH][KB + 8];       // transposed: [channel][key]
    __shared__ bf16 Ps[8][16][KB + 8];     // wave-private P staging
    int b = blockIdx.y, tid = threadIdx.x;
    int lane = tid & 31, wave = tid >> 5;
    int half = lane >> 4, l16 = lane & 15;
    const bf16* Q  = qws + (size_t)b * NPIX * CCH;
    const bf16* Kg = kws + (size_t)b * NPIX * CCH;
    const bf16* Vg = vws + (size_t)b * NPIX * CCH;
    int qbase = blockIdx.x * 128 + wave * 16;

    v16bf qf[4];
    #pragma unroll
    for (int i = 0; i < 4; i++) qf[i] = loadA(Q, CCH, qbase, i * 32, lane);

    v8f O[8];
    for (int t = 0; t < 8; t++) for (int r = 0; r < 8; r++) O[t][r] = 0.f;
    float m[8], l[8];
    for (int r = 0; r < 8; r++) { m[r] = -1e30f; l[r] = 0.f; }

    int key = tid >> 2;            // 0..63
    int ch  = (tid & 3) * 32;      // 0,32,64,96 (64B per thread per matrix)

    for (int kb = 0; kb < NPIX; kb += KB) {
        // ---- stage K block via async DMA to LDS (ASYNCcnt path) ----
        async_copy64(Kg + (size_t)(kb + key) * CCH + ch, &Kblk[key][ch]);

        // ---- stage V block transposed (needs layout change) ----
        {
            const bf16* vs = Vg + (size_t)(kb + key) * CCH + ch;
            #pragma unroll
            for (int jj = 0; jj < 4; jj++) {
                v8bf vv = *reinterpret_cast<const v8bf*>(vs + jj * 8);
                #pragma unroll
                for (int j = 0; j < 8; j++) Vt[ch + jj * 8 + j][key] = vv[j];
            }
        }
        wait_async0();
        __syncthreads();
        if (kb + KB < NPIX) {      // pull next block toward L2/L1
            __builtin_prefetch(Kg + (size_t)(kb + KB + key) * CCH + ch, 0, 1);
            __builtin_prefetch(Vg + (size_t)(kb + KB + key) * CCH + ch, 0, 1);
        }

        // ---- S = Q * K^T : four 16x16 tiles over 64 keys ----
        v8f S[4];
        #pragma unroll
        for (int j = 0; j < 4; j++) for (int r = 0; r < 8; r++) S[j][r] = 0.f;
        #pragma unroll
        for (int i = 0; i < 4; i++) {
            #pragma unroll
            for (int j = 0; j < 4; j++) {
                v16bf bkf = loadB(&Kblk[0][0], CCH + 8, j * 16, i * 32, lane);
                S[j] = wmma_bf16(qf[i], bkf, S[j]);
            }
        }

        // ---- online softmax; row (r + 8*half) values live in elem r across a
        //      16-lane half, so xor-masks 1,2,4,8 reduce both halves at once ----
        float alpha[8];
        #pragma unroll
        for (int r = 0; r < 8; r++) {
            float v = fmaxf(fmaxf(S[0][r], S[1][r]), fmaxf(S[2][r], S[3][r]));
            for (int d = 8; d >= 1; d >>= 1) v = fmaxf(v, __shfl_xor(v, d));
            float mn = fmaxf(m[r], v);
            alpha[r] = __expf(m[r] - mn);
            m[r] = mn;
            float rs = 0.f;
            #pragma unroll
            for (int j = 0; j < 4; j++) { S[j][r] = __expf(S[j][r] - mn); rs += S[j][r]; }
            for (int d = 8; d >= 1; d >>= 1) rs += __shfl_xor(rs, d);
            l[r] = l[r] * alpha[r] + rs;
        }
        #pragma unroll
        for (int t = 0; t < 8; t++)
            for (int r = 0; r < 8; r++) O[t][r] *= alpha[r];

        // ---- restage P into A-fragment layout via wave-private LDS ----
        #pragma unroll
        for (int r = 0; r < 8; r++) {
            int row = r + half * 8;
            #pragma unroll
            for (int j = 0; j < 4; j++)
                Ps[wave][row][j * 16 + l16] = f2bf(S[j][r]);
        }
        v16bf pf0 = loadA(&Ps[wave][0][0], KB + 8, 0, 0,  lane);
        v16bf pf1 = loadA(&Ps[wave][0][0], KB + 8, 0, 32, lane);

        // ---- O += P * V ----
        #pragma unroll
        for (int t = 0; t < 8; t++) {
            O[t] = wmma_bf16(pf0, loadB(&Vt[0][0], KB + 8, t * 16, 0,  lane), O[t]);
            O[t] = wmma_bf16(pf1, loadB(&Vt[0][0], KB + 8, t * 16, 32, lane), O[t]);
        }
        __syncthreads();
    }

    // ---- epilogue: normalize rows, store attn output (b,n,c) bf16 ----
    bf16* Aout = aws + (size_t)b * NPIX * CCH;
    float inv[8];
    for (int r = 0; r < 8; r++) inv[r] = 1.f / l[r];
    for (int t = 0; t < 8; t++) {
        for (int r = 0; r < 8; r++) {
            int q = qbase + r + half * 8;
            int d = t * 16 + l16;
            Aout[(size_t)q * CCH + d] = f2bf(O[t][r] * inv[r]);
        }
    }
}

// ============ kernel 4: proj conv1x1 + bias + residual (WMMA) ============
__global__ __launch_bounds__(256) void proj_kernel(const bf16* __restrict__ aws,
                                                   const bf16* __restrict__ wpbf,
                                                   const float* __restrict__ bp,
                                                   const float* __restrict__ x,
                                                   float* __restrict__ out) {
    int b = blockIdx.y, pblk = blockIdx.x, tid = threadIdx.x;
    int lane = tid & 31, wave = tid >> 5;
    int half = lane >> 4, l16 = lane & 15;
    int pbase = pblk * 128;
    const bf16* A = aws + (size_t)b * NPIX * CCH;

    v16bf wf[4];
    #pragma unroll
    for (int i = 0; i < 4; i++) wf[i] = loadA(wpbf, CCH, wave * 16, i * 32, lane);

    for (int pt = 0; pt < 8; pt++) {
        v8f acc; for (int r = 0; r < 8; r++) acc[r] = 0.f;
        #pragma unroll
        for (int i = 0; i < 4; i++)
            acc = wmma_bf16(wf[i], loadB(A, CCH, pbase + pt * 16, i * 32, lane), acc);
        int p = pbase + pt * 16 + l16;
        #pragma unroll
        for (int r = 0; r < 8; r++) {
            int o = wave * 16 + r + half * 8;
            size_t idx = ((size_t)b * CCH + o) * NPIX + p;
            out[idx] = x[idx] + acc[r] + bp[o];
        }
    }
}

// ============ launch ============
extern "C" void kernel_launch(void* const* d_in, const int* in_sizes, int n_in,
                              void* d_out, int out_size, void* d_ws, size_t ws_size,
                              hipStream_t stream) {
    const float* x     = (const float*)d_in[0];
    const float* gamma = (const float*)d_in[1];
    const float* beta  = (const float*)d_in[2];
    const float* wq    = (const float*)d_in[3];
    const float* bq    = (const float*)d_in[4];
    const float* wk    = (const float*)d_in[5];
    const float* bk    = (const float*)d_in[6];
    const float* wv    = (const float*)d_in[7];
    const float* bv    = (const float*)d_in[8];
    const float* wp    = (const float*)d_in[9];
    const float* bp    = (const float*)d_in[10];
    float* out = (float*)d_out;

    // workspace layout (all 16B aligned)
    char* ws = (char*)d_ws;
    bf16*  wbf   = (bf16*)ws;                                   // 4*16384 bf16
    float* stats = (float*)(ws + 131072);                       // 32*2 f32
    size_t mat = (size_t)NB * NPIX * CCH * sizeof(bf16);        // 4 MiB each
    bf16* qws = (bf16*)(ws + 131328);
    bf16* kws = (bf16*)(ws + 131328 + mat);
    bf16* vws = (bf16*)(ws + 131328 + 2 * mat);
    bf16* aws = (bf16*)(ws + 131328 + 3 * mat);

    wcvt_kernel<<<256, 256, 0, stream>>>(wq, wk, wv, wp, wbf);
    gn_stats_kernel<<<dim3(8, NB), 256, 0, stream>>>(x, stats);
    qkv_kernel<<<dim3(32, NB), 256, 0, stream>>>(x, gamma, beta, wbf, bq, bk, bv,
                                                 stats, qws, kws, vws);
    attn_kernel<<<dim3(32, NB), 256, 0, stream>>>(qws, kws, vws, aws);
    proj_kernel<<<dim3(32, NB), 256, 0, stream>>>(aws, wbf + 3 * 16384, bp, x, out);
}
